// FasterRCNNVGG16_4363686773100
// MI455X (gfx1250) — matrix operand order repaired and checked by reading, not compiled, not run on hardware
//
#include <hip/hip_runtime.h>
#include <hip/hip_bf16.h>
#include <math.h>

// ---------------------------------------------------------------------------
// Types for WMMA fragments (avoid HIP struct vector types inside unions).
// ---------------------------------------------------------------------------
typedef __attribute__((ext_vector_type(16))) __bf16        v16bf;
typedef __attribute__((ext_vector_type(8)))  float         v8f;
typedef __attribute__((ext_vector_type(4)))  unsigned int  u32x4;

union FragU {
    v16bf v;
    u32x4 q[2];
    unsigned int d[8];
    unsigned short s[16];
};

__device__ __forceinline__ unsigned short f32_to_bf16(float f) {
    unsigned u = __float_as_uint(f);
    unsigned r = u + 0x7fffu + ((u >> 16) & 1u);   // round-to-nearest-even
    return (unsigned short)(r >> 16);
}
__device__ __forceinline__ float bf16_to_f32(unsigned short s) {
    return __uint_as_float(((unsigned)s) << 16);
}

// pack the high halves (bf16, truncated) of two fp32 bit patterns: 1 VALU op
__device__ __forceinline__ unsigned pk_hi16(unsigned lo, unsigned hi) {
#if __has_builtin(__builtin_amdgcn_perm)
    return __builtin_amdgcn_perm(hi, lo, 0x07060302u);   // v_perm_b32
#else
    return (lo >> 16) | (hi & 0xffff0000u);              // v_lshr + v_and_or
#endif
}

// ---------------------------------------------------------------------------
// Problem constants
// ---------------------------------------------------------------------------
#define N_CLASS   11
#define R_NUM     512
#define FEAT_C    512
#define FEAT_H    37
#define FEAT_W    50
#define POOL_K    25088          // 512*7*7
#define FC_N      4096
#define IMG_H_F   600.0f
#define IMG_W_F   800.0f
#define NMS_T     0.3f
#define SCORE_T   0.05f

#define RAWSTRIDE 68             // dwords per k-row of raw fp32 tile (64 + pad)
#define BFSTRIDE  20             // dwords per n-column of bf16 tile (16 + pad)

// ---------------------------------------------------------------------------
// 1) ROI max pool  ->  bf16 A-matrix [512 x 25088]
// ---------------------------------------------------------------------------
__global__ void roi_pool_kernel(const float* __restrict__ feat,
                                const float* __restrict__ rois,
                                const int*   __restrict__ roi_idx,
                                unsigned short* __restrict__ A)
{
    int idx = blockIdx.x * blockDim.x + threadIdx.x;
    const int total = R_NUM * FEAT_C * 49;
    if (idx >= total) return;

    int bin = idx % 49;
    int pw  = bin % 7;
    int ph  = bin / 7;
    int t   = idx / 49;
    int c   = t % FEAT_C;
    int r   = t / FEAT_C;

    float y1 = rois[r * 4 + 0];
    float x1 = rois[r * 4 + 1];
    float y2 = rois[r * 4 + 2];
    float x2 = rois[r * 4 + 3];

    const float SS = 1.0f / 16.0f;
    float sw = rintf(x1 * SS);
    float ew = rintf(x2 * SS);
    float sh = rintf(y1 * SS);
    float eh = rintf(y2 * SS);
    float rw = fmaxf(ew - sw + 1.0f, 1.0f);
    float rh = fmaxf(eh - sh + 1.0f, 1.0f);

    float hs_f = fminf(fmaxf(floorf((float)ph       * rh / 7.0f) + sh, 0.0f), (float)FEAT_H);
    float he_f = fminf(fmaxf(ceilf ((float)(ph + 1) * rh / 7.0f) + sh, 0.0f), (float)FEAT_H);
    float ws_f = fminf(fmaxf(floorf((float)pw       * rw / 7.0f) + sw, 0.0f), (float)FEAT_W);
    float we_f = fminf(fmaxf(ceilf ((float)(pw + 1) * rw / 7.0f) + sw, 0.0f), (float)FEAT_W);

    int hs = (int)hs_f, he = (int)he_f, ws = (int)ws_f, we = (int)we_f;

    float val = 0.0f;
    if (he > hs && we > ws) {
        int b = roi_idx[r];
        const float* f = feat + ((size_t)b * FEAT_C + c) * (FEAT_H * FEAT_W);
        float m = -INFINITY;
        for (int y = hs; y < he; ++y) {
            const float* row = f + y * FEAT_W;
            for (int x = ws; x < we; ++x)
                m = fmaxf(m, row[x]);
        }
        val = m;
    }
    A[(size_t)r * POOL_K + c * 49 + ph * 7 + pw] = f32_to_bf16(val);
}

// ---------------------------------------------------------------------------
// 2) bf16 WMMA GEMM:  C = relu?( A[M,K](bf16) * W[K,N](f32) + bias )
//    Block tile 128x64, 128 threads (4 waves), wave tile 32x64 (8 frags).
//    Pipeline per 32-k slab:
//      - GLOBAL_LOAD_ASYNC_TO_LDS_B128 streams the fp32 weight slab into a
//        ping-pong raw [k][n] LDS tile (4 async ops/thread, ASYNCcnt).
//      - after s_wait_asynccnt, each wave packs ONLY its own async-written
//        (k,k+1) row pairs into a shared bf16 [n][k/2] tile (1 VALU/dword).
//      - fragments are then pure ds_load_b128, feeding 8 WMMAs per wave.
// ---------------------------------------------------------------------------
__global__ void __launch_bounds__(128)
gemm_bf16_kernel(const unsigned short* __restrict__ A,
                 const float* __restrict__ W,
                 const float* __restrict__ bias,
                 unsigned short* __restrict__ C,
                 int M, int N, int K, int do_relu)
{
    __shared__ unsigned raw[2][32 * RAWSTRIDE];   // fp32 bits, [k][n]+pad
    __shared__ unsigned bfs[64 * BFSTRIDE];       // bf16 pairs, [n][k/2]+pad

    const int tid  = threadIdx.x;
    const int wave = tid >> 5;               // 0..3
    const int lane = tid & 31;
    const int half = lane >> 4;              // 0..1
    const int l15  = lane & 15;

    const int nBlk = blockIdx.x * 64;
    const int mBlk = blockIdx.y * 128;
    const int m0   = mBlk + wave * 32;       // wave's 32-row strip

    // async staging: 512 b128-chunks per slab, 4 per thread (chunk = tid+128q)
    const int kl0  = tid >> 4;               // k row of chunk q: kl0 + 8q
    const int nq4  = (tid & 15) * 4;         // n offset of all 4 chunks

    // convert-pass ownership: wave w owns row pairs k0 = 2w + 8p, p=0..3
    const int cp   = lane >> 3;              // 0..3
    const int cn8  = (lane & 7) * 8;         // 8 n-columns per lane
    const int ck0  = 2 * wave + 8 * cp;
    const int ckp  = ck0 >> 1;               // bf16-pair column index

    v8f acc[8];
    #pragma unroll
    for (int i = 0; i < 8; ++i) acc[i] = v8f{0, 0, 0, 0, 0, 0, 0, 0};

    const unsigned short* Arow0 = A + (size_t)(m0 + l15)      * K;
    const unsigned short* Arow1 = A + (size_t)(m0 + 16 + l15) * K;

    // ---- issue async stage for k-slab `ko` into buffer `buf` ----
    auto issue_stage = [&](int ko, int buf) {
        const float* sbase = W + (size_t)ko * N + nBlk;   // uniform -> SGPR pair
        #pragma unroll
        for (int q = 0; q < 4; ++q) {
            int kl = kl0 + 8 * q;
            unsigned lds = (unsigned)(uintptr_t)&raw[buf][kl * RAWSTRIDE + nq4];
            int voff = (kl * N + nq4) * 4;                // byte offset, lane-varying
            asm volatile("global_load_async_to_lds_b128 %0, %1, %2"
                         :: "v"(lds), "v"(voff), "s"(sbase) : "memory");
        }
    };

    issue_stage(0, 0);

    int ib = 0;
    for (int ko = 0; ko < K; ko += 32, ib ^= 1) {
        const bool has_next = (ko + 32) < K;
        if (has_next) {
            issue_stage(ko + 32, ib ^ 1);
            if (ko + 96 < K)   // deep L2 prefetch, 3 slabs ahead
                __builtin_prefetch(W + (size_t)(ko + 96 + kl0) * N + nBlk + nq4, 0, 1);
            asm volatile("s_wait_asynccnt 0x4" ::: "memory");   // slab ko landed
        } else {
            asm volatile("s_wait_asynccnt 0x0" ::: "memory");
        }

        // ---- convert own (k,k+1) row pairs: fp32 -> packed bf16 [n][k/2] ----
        {
            const unsigned* r0 = &raw[ib][ ck0      * RAWSTRIDE + cn8];
            const unsigned* r1 = &raw[ib][(ck0 + 1) * RAWSTRIDE + cn8];
            u32x4 aL = *(const u32x4*)(r0);
            u32x4 aH = *(const u32x4*)(r0 + 4);
            u32x4 bL = *(const u32x4*)(r1);
            u32x4 bH = *(const u32x4*)(r1 + 4);
            #pragma unroll
            for (int j = 0; j < 4; ++j)
                bfs[(cn8 + j) * BFSTRIDE + ckp] = pk_hi16(aL[j], bL[j]);
            #pragma unroll
            for (int j = 0; j < 4; ++j)
                bfs[(cn8 + 4 + j) * BFSTRIDE + ckp] = pk_hi16(aH[j], bH[j]);
        }
        __syncthreads();

        // ---- A fragments (bf16, direct from global) ----
        FragU a0, a1;
        a0.q[0] = *(const u32x4*)(Arow0 + ko + half * 8);
        a0.q[1] = *(const u32x4*)(Arow0 + ko + 16 + half * 8);
        a1.q[0] = *(const u32x4*)(Arow1 + ko + half * 8);
        a1.q[1] = *(const u32x4*)(Arow1 + ko + 16 + half * 8);

        // ---- 4 B fragments: pure ds_load_b128, each reused by 2 WMMAs ----
        #pragma unroll
        for (int t = 0; t < 4; ++t) {
            const unsigned* bp = &bfs[(t * 16 + l15) * BFSTRIDE + half * 4];
            FragU b;
            b.q[0] = *(const u32x4*)(bp);
            b.q[1] = *(const u32x4*)(bp + 8);

            acc[t]     = __builtin_amdgcn_wmma_f32_16x16x32_bf16(false, a0.v, false, b.v,
                                                                 (short)0, acc[t],     false, false);
            acc[t + 4] = __builtin_amdgcn_wmma_f32_16x16x32_bf16(false, a1.v, false, b.v,
                                                                 (short)0, acc[t + 4], false, false);
        }
        __syncthreads();
    }

    // ---- epilogue: bias (+ReLU), convert to bf16, scatter-store ----
    #pragma unroll
    for (int s = 0; s < 2; ++s) {
        #pragma unroll
        for (int t = 0; t < 4; ++t) {
            int n = nBlk + t * 16 + l15;
            float bb = bias[n];
            #pragma unroll
            for (int v = 0; v < 8; ++v) {
                int m = m0 + s * 16 + half * 8 + v;
                float x = acc[s * 4 + t][v] + bb;
                if (do_relu) x = fmaxf(x, 0.0f);
                C[(size_t)m * N + n] = f32_to_bf16(x);
            }
        }
    }
}

// ---------------------------------------------------------------------------
// 3) heads: loc = fc7 @ Wloc + bloc (512x44), score = fc7 @ Wscore + bscore
// ---------------------------------------------------------------------------
__global__ void heads_kernel(const unsigned short* __restrict__ fc7,
                             const float* __restrict__ Wloc,
                             const float* __restrict__ bloc,
                             const float* __restrict__ Wsc,
                             const float* __restrict__ bsc,
                             float* __restrict__ loc,
                             float* __restrict__ score)
{
    int r = blockIdx.x;
    int j = threadIdx.x;
    const unsigned short* row = fc7 + (size_t)r * FC_N;

    if (j < N_CLASS * 4) {
        float acc = bloc[j];
        for (int k = 0; k < FC_N; ++k)
            acc += bf16_to_f32(row[k]) * Wloc[(size_t)k * (N_CLASS * 4) + j];
        loc[r * (N_CLASS * 4) + j] = acc;
    } else if (j < N_CLASS * 4 + N_CLASS) {
        int s = j - N_CLASS * 4;
        float acc = bsc[s];
        for (int k = 0; k < FC_N; ++k)
            acc += bf16_to_f32(row[k]) * Wsc[(size_t)k * N_CLASS + s];
        score[r * N_CLASS + s] = acc;
    }
}

// ---------------------------------------------------------------------------
// 4) postprocess: softmax -> prob, loc -> loc2bbox -> clip -> cls_bbox
//    d_out: [0,22528) cls_bbox | [22528,28160) prob | [28160,33280) keep
// ---------------------------------------------------------------------------
__global__ void post_kernel(const float* __restrict__ rois,
                            const float* __restrict__ loc,
                            const float* __restrict__ score,
                            float* __restrict__ out)
{
    int r = blockIdx.x * blockDim.x + threadIdx.x;
    if (r >= R_NUM) return;

    float mx = -INFINITY;
    for (int s = 0; s < N_CLASS; ++s) mx = fmaxf(mx, score[r * N_CLASS + s]);
    float sum = 0.0f, e[N_CLASS];
    for (int s = 0; s < N_CLASS; ++s) { e[s] = expf(score[r * N_CLASS + s] - mx); sum += e[s]; }
    float inv = 1.0f / sum;
    for (int s = 0; s < N_CLASS; ++s) out[R_NUM * 44 + r * N_CLASS + s] = e[s] * inv;

    float y1 = rois[r * 4 + 0], x1 = rois[r * 4 + 1];
    float y2 = rois[r * 4 + 2], x2 = rois[r * 4 + 3];
    float sh = y2 - y1, sw = x2 - x1;
    float cy = y1 + 0.5f * sh, cx = x1 + 0.5f * sw;

    for (int c = 0; c < N_CLASS; ++c) {
        float dy = loc[r * 44 + c * 4 + 0] * 0.1f;
        float dx = loc[r * 44 + c * 4 + 1] * 0.1f;
        float dh = loc[r * 44 + c * 4 + 2] * 0.2f;
        float dw = loc[r * 44 + c * 4 + 3] * 0.2f;
        float ncy = dy * sh + cy;
        float ncx = dx * sw + cx;
        float nh  = expf(dh) * sh;
        float nw  = expf(dw) * sw;
        out[r * 44 + c * 4 + 0] = fminf(fmaxf(ncy - 0.5f * nh, 0.0f), IMG_H_F);
        out[r * 44 + c * 4 + 1] = fminf(fmaxf(ncx - 0.5f * nw, 0.0f), IMG_W_F);
        out[r * 44 + c * 4 + 2] = fminf(fmaxf(ncy + 0.5f * nh, 0.0f), IMG_H_F);
        out[r * 44 + c * 4 + 3] = fminf(fmaxf(ncx + 0.5f * nw, 0.0f), IMG_W_F);
    }
}

// ---------------------------------------------------------------------------
// 5) per-class greedy NMS (classes 1..10). One block per class, 512 threads.
// ---------------------------------------------------------------------------
__global__ void __launch_bounds__(512)
nms_kernel(float* __restrict__ out)
{
    __shared__ float key[R_NUM];
    __shared__ int   sidx[R_NUM];
    __shared__ float bx[R_NUM * 4];
    __shared__ float area[R_NUM];
    __shared__ int   keep[R_NUM];

    const int cls = blockIdx.x + 1;
    const int t   = threadIdx.x;
    const float* bbox = out;
    const float* prob = out + R_NUM * 44;
    float* keep_out   = out + R_NUM * 44 + R_NUM * N_CLASS;

    float sc = prob[t * N_CLASS + cls];
    int valid = sc > SCORE_T;
    key[t] = valid ? -sc : INFINITY;
    __syncthreads();

    float myk = key[t];
    int rank = 0;
    for (int j = 0; j < R_NUM; ++j) {
        float kj = key[j];
        rank += (kj < myk) || (kj == myk && j < t);
    }
    sidx[rank] = t;
    __syncthreads();

    int orig = sidx[t];
    float ymin = bbox[orig * 44 + cls * 4 + 0];
    float xmin = bbox[orig * 44 + cls * 4 + 1];
    float ymax = bbox[orig * 44 + cls * 4 + 2];
    float xmax = bbox[orig * 44 + cls * 4 + 3];
    bx[t * 4 + 0] = ymin; bx[t * 4 + 1] = xmin;
    bx[t * 4 + 2] = ymax; bx[t * 4 + 3] = xmax;
    area[t] = (ymax - ymin) * (xmax - xmin);
    keep[t] = (prob[orig * N_CLASS + cls] > SCORE_T) ? 1 : 0;
    __syncthreads();

    for (int i = 0; i < R_NUM; ++i) {
        __syncthreads();
        if (t > i && keep[i] && keep[t]) {
            float ty = fmaxf(bx[i * 4 + 0], bx[t * 4 + 0]);
            float tx = fmaxf(bx[i * 4 + 1], bx[t * 4 + 1]);
            float by = fminf(bx[i * 4 + 2], bx[t * 4 + 2]);
            float bxx = fminf(bx[i * 4 + 3], bx[t * 4 + 3]);
            float hh = fmaxf(by - ty, 0.0f);
            float ww = fmaxf(bxx - tx, 0.0f);
            float inter = hh * ww;
            float iou = inter / (area[i] + area[t] - inter);
            if (iou > NMS_T) keep[t] = 0;
        }
    }
    __syncthreads();
    keep_out[(size_t)blockIdx.x * R_NUM + orig] = keep[t] ? 1.0f : 0.0f;
}

// ---------------------------------------------------------------------------
// launch
// ---------------------------------------------------------------------------
extern "C" void kernel_launch(void* const* d_in, const int* in_sizes, int n_in,
                              void* d_out, int out_size, void* d_ws, size_t ws_size,
                              hipStream_t stream)
{
    const float* h      = (const float*)d_in[0];
    const float* rois   = (const float*)d_in[1];
    const int*   ridx   = (const int*)  d_in[2];
    const float* W1     = (const float*)d_in[3];
    const float* b1     = (const float*)d_in[4];
    const float* W2     = (const float*)d_in[5];
    const float* b2     = (const float*)d_in[6];
    const float* Wloc   = (const float*)d_in[7];
    const float* bloc   = (const float*)d_in[8];
    const float* Wscore = (const float*)d_in[9];
    const float* bscore = (const float*)d_in[10];
    float* out = (float*)d_out;

    char* ws = (char*)d_ws;
    unsigned short* A1   = (unsigned short*)(ws);                    // 512*25088 bf16
    unsigned short* fc   = (unsigned short*)(ws + 25690112);         // 512*4096 bf16
    unsigned short* fc7  = (unsigned short*)(ws + 25690112 + 4194304);
    float*          loc  = (float*)(ws + 25690112 + 2 * 4194304);    // 512*44 f32
    float*          scr  = (float*)(ws + 25690112 + 2 * 4194304 + 90112); // 512*11

    {
        int total = R_NUM * FEAT_C * 49;
        roi_pool_kernel<<<(total + 255) / 256, 256, 0, stream>>>(h, rois, ridx, A1);
    }
    // FC1: [512,25088] x [25088,4096] + b1, ReLU
    gemm_bf16_kernel<<<dim3(FC_N / 64, R_NUM / 128), 128, 0, stream>>>(
        A1, W1, b1, fc, R_NUM, FC_N, POOL_K, 1);
    // FC2: [512,4096] x [4096,4096] + b2, ReLU
    gemm_bf16_kernel<<<dim3(FC_N / 64, R_NUM / 128), 128, 0, stream>>>(
        fc, W2, b2, fc7, R_NUM, FC_N, FC_N, 1);
    heads_kernel<<<R_NUM, 64, 0, stream>>>(fc7, Wloc, bloc, Wscore, bscore, loc, scr);
    post_kernel<<<(R_NUM + 255) / 256, 256, 0, stream>>>(rois, loc, scr, out);
    nms_kernel<<<N_CLASS - 1, R_NUM, 0, stream>>>(out);
}
